// MetaLearningRALECGNN_58025008169380
// MI455X (gfx1250) — compile-verified, matrix operand order checked by reading, not run on hardware
//
#include <hip/hip_runtime.h>
#include <hip/hip_bf16.h>

#define NN 1024
#define FF 128
#define MM 256
#define PP 523776   // N*(N-1)/2
#define BN_EPS 1e-5f

typedef __attribute__((ext_vector_type(16))) _Float16 v16h;
typedef __attribute__((ext_vector_type(8)))  _Float16 v8h;
typedef __attribute__((ext_vector_type(8)))  float    v8f;

// ---------------------------------------------------------------------------
// Kernel 1: fold BatchNorm + ctx into small tensors.
//   cterm[m] = b1[m] + sum_k ctx[k]*W1[m][256+k]                (stored f16)
//   s[k] = gamma*rsqrt(var+eps); t[k] = beta - s*mean
//   W2frag: B-matrix (256x16, cols 8..15 zero) pre-swizzled into the f16
//           16x32 WMMA B-fragment layout: [kt 0..7][lane 0..31][elem 0..15]
//   b2p[e] = b2[e] + sum_k W2[e][k]*t[k]  (padded 8..15 with 0)
// ---------------------------------------------------------------------------
__global__ void fold_kernel(const float* __restrict__ ctx, const float* __restrict__ W1,
                            const float* __restrict__ b1, const float* __restrict__ gamma,
                            const float* __restrict__ beta, const float* __restrict__ rmean,
                            const float* __restrict__ rvar, const float* __restrict__ W2,
                            const float* __restrict__ b2,
                            _Float16* __restrict__ cterm_h, float* __restrict__ b2p,
                            _Float16* __restrict__ W2frag) {
    int t = threadIdx.x;                     // 256 threads
    // cterm (f16)
    float c = b1[t];
    #pragma unroll
    for (int k = 0; k < 4; ++k) c += ctx[k] * W1[t * 260 + 256 + k];
    cterm_h[t] = (_Float16)c;

    // B fragment pre-swizzle: thread t <-> (kt, lane)
    int lane = t & 31;
    int kt = t >> 5;
    int khalf = lane >> 4;
    int ncol = lane & 15;
    #pragma unroll
    for (int r = 0; r < 8; ++r) {
        #pragma unroll
        for (int s = 0; s < 2; ++s) {
            int kl = ((r >= 4) ? 16 : 0) + khalf * 8 + (r & 3) * 2 + s;
            int k = kt * 32 + kl;
            float v = 0.0f;
            if (ncol < 8) {
                float sk = gamma[k] * rsqrtf(rvar[k] + BN_EPS);
                v = W2[ncol * 256 + k] * sk;
            }
            W2frag[t * 16 + r * 2 + s] = (_Float16)v;
        }
    }

    if (t < 16) {
        float acc = 0.0f;
        if (t < 8) {
            acc = b2[t];
            for (int k = 0; k < 256; ++k) {
                float sk = gamma[k] * rsqrtf(rvar[k] + BN_EPS);
                float tk = beta[k] - sk * rmean[k];
                acc += W2[t * 256 + k] * tk;
            }
        }
        b2p[t] = acc;
    }
}

// ---------------------------------------------------------------------------
// Kernel 2: pa = x @ W1[:, :F].T ; pb = x @ W1[:, F:2F].T
// f32 accumulate (67 MFLOP total), results stored as f16 for the pair phase.
// W1 rows are 260 floats = 1040B (16B-aligned), so float4 loads are legal.
// ---------------------------------------------------------------------------
__global__ __launch_bounds__(256) void pp_kernel(const float* __restrict__ x,
                                                 const float* __restrict__ W1,
                                                 _Float16* __restrict__ pa,
                                                 _Float16* __restrict__ pb) {
    __shared__ __align__(16) float xs[FF];
    int n = blockIdx.x;
    int m = threadIdx.x;
    if (m < FF) xs[m] = x[n * FF + m];
    __syncthreads();
    const float4* wa4 = (const float4*)(W1 + m * 260);
    const float4* wb4 = (const float4*)(W1 + m * 260 + FF);
    const float4* xs4 = (const float4*)xs;
    float a = 0.f, b = 0.f;
    #pragma unroll 8
    for (int k = 0; k < FF / 4; ++k) {
        float4 xv = xs4[k], wa = wa4[k], wb = wb4[k];
        a = fmaf(xv.x, wa.x, fmaf(xv.y, wa.y, fmaf(xv.z, wa.z, fmaf(xv.w, wa.w, a))));
        b = fmaf(xv.x, wb.x, fmaf(xv.y, wb.y, fmaf(xv.z, wb.z, fmaf(xv.w, wb.w, b))));
    }
    pa[n * MM + m] = (_Float16)a;
    pb[n * MM + m] = (_Float16)b;
}

// ---------------------------------------------------------------------------
// Kernel 3: per-node unbiased std over feature dim (wave32 per node).
// ---------------------------------------------------------------------------
__global__ void stress_kernel(const float* __restrict__ x, float* __restrict__ stress) {
    int wave = threadIdx.x >> 5;
    int lane = threadIdx.x & 31;
    int n = blockIdx.x * 4 + wave;
    float s = 0.f, ss = 0.f;
    #pragma unroll
    for (int k = lane; k < FF; k += 32) {
        float v = x[n * FF + k];
        s += v;
        ss = fmaf(v, v, ss);
    }
    #pragma unroll
    for (int m = 16; m >= 1; m >>= 1) {
        s  += __shfl_xor(s,  m, 32);
        ss += __shfl_xor(ss, m, 32);
    }
    if (lane == 0) {
        float mean = s * (1.0f / FF);
        float var = (ss - (float)FF * mean * mean) * (1.0f / (FF - 1));
        stress[n] = sqrtf(fmaxf(var, 0.f));
    }
}

// ---------------------------------------------------------------------------
// Kernel 4: mean of stress + stressed flags (1 block of 1024).
// ---------------------------------------------------------------------------
__global__ __launch_bounds__(1024) void flags_kernel(const float* __restrict__ stress,
                                                     float* __restrict__ flags) {
    __shared__ float red[1024];
    int t = threadIdx.x;
    float v = stress[t];
    red[t] = v;
    __syncthreads();
    for (int s = 512; s > 0; s >>= 1) {
        if (t < s) red[t] += red[t + s];
        __syncthreads();
    }
    float mean = red[0] * (1.0f / NN);
    flags[t] = (v > mean) ? 1.0f : 0.0f;
}

// ---------------------------------------------------------------------------
// Pair-index decode: linear p -> (i, j), i < j, triu order.
// ---------------------------------------------------------------------------
__device__ __forceinline__ void decode_pair(int p, int& i, int& j) {
    float disc = (2.f * NN - 1.f) * (2.f * NN - 1.f) - 8.f * (float)p;
    int ii = (int)((2.f * NN - 1.f - sqrtf(fmaxf(disc, 0.f))) * 0.5f);
    ii = max(0, min(ii, NN - 2));
    while (ii > 0 && (ii * (2 * NN - 1 - ii)) / 2 > p) --ii;
    while (((ii + 1) * (2 * NN - 1 - (ii + 1))) / 2 <= p) ++ii;
    i = ii;
    j = ii + 1 + (p - (ii * (2 * NN - 1 - ii)) / 2);
}

// Build the f16 16x32 A fragment for one pair-row: two contiguous 8-half
// K-runs (k0..k0+7 and k0+16..k0+23); 16B vector loads, packed f16 add/relu.
__device__ __forceinline__ v16h build_a(const _Float16* __restrict__ par,
                                        const _Float16* __restrict__ pbr,
                                        const _Float16* __restrict__ ct, int k0) {
    v8h a0 = *(const v8h*)(par + k0);
    v8h a1 = *(const v8h*)(par + k0 + 16);
    v8h b0 = *(const v8h*)(pbr + k0);
    v8h b1 = *(const v8h*)(pbr + k0 + 16);
    v8h c0 = *(const v8h*)(ct + k0);
    v8h c1 = *(const v8h*)(ct + k0 + 16);
    v8h z = {};
    v8h h0 = __builtin_elementwise_max((v8h)(a0 + b0 + c0), z);   // v_pk_add/max_f16
    v8h h1 = __builtin_elementwise_max((v8h)(a1 + b1 + c1), z);
    return __builtin_shufflevector(h0, h1, 0, 1, 2, 3, 4, 5, 6, 7,
                                   8, 9, 10, 11, 12, 13, 14, 15);
}

// Epilogue for one accumulator tile half: e_n = relu(D+b2')*w3, reduce over n.
__device__ __forceinline__ void epilogue(v8f c, int lane, float b2n, float w3n,
                                         float b3, int p, int i, int j,
                                         const float* __restrict__ flags,
                                         float* __restrict__ scores,
                                         float* __restrict__ maskOut) {
    float tsum[8];
    #pragma unroll
    for (int d = 0; d < 8; ++d) {
        float e = fmaxf(c[d] + b2n, 0.0f) * w3n;
        e += __shfl_xor(e, 1, 32);
        e += __shfl_xor(e, 2, 32);
        e += __shfl_xor(e, 4, 32);
        e += __shfl_xor(e, 8, 32);
        tsum[d] = e;
    }
    // lane L<8 owns m=L (M=d group), lane 24..31 owns m=L-16 (M=d+8 group)
    if (lane < 8 || lane >= 24) {
        int d = lane & 7;
        float sc = 1.0f / (1.0f + __expf(-(tsum[d] + b3)));
        scores[p] = sc;
        maskOut[p] = (sc > 0.5f) ? (flags[i] * flags[j]) : 0.0f;
    }
}

// ---------------------------------------------------------------------------
// Kernel 5: pair scorer. One wave per 32-pair tile: two A fragments share one
// pre-swizzled B fragment; 16x V_WMMA_F32_16X16X32_F16 per wave (unrolled).
// ---------------------------------------------------------------------------
__global__ __launch_bounds__(128) void pair_kernel(
    const _Float16* __restrict__ pa, const _Float16* __restrict__ pb,
    const _Float16* __restrict__ cterm_g, const _Float16* __restrict__ W2frag_g,
    const float* __restrict__ b2p, const float* __restrict__ W3,
    const float* __restrict__ b3p, const float* __restrict__ flags,
    float* __restrict__ scores, float* __restrict__ maskOut) {
    __shared__ __align__(16) _Float16 cterm[MM];
    __shared__ __align__(32) _Float16 W2f[8 * 32 * 16];

    int tid = threadIdx.x;
    {   // stage constants to LDS (vectorized)
        const float4* s1 = (const float4*)cterm_g;     // 512B = 32 float4
        float4* d1 = (float4*)cterm;
        for (int k = tid; k < 32; k += 128) d1[k] = s1[k];
        const float4* s2 = (const float4*)W2frag_g;    // 8192B = 512 float4
        float4* d2 = (float4*)W2f;
        for (int k = tid; k < 512; k += 128) d2[k] = s2[k];
    }
    __syncthreads();

    int wave = tid >> 5;
    int lane = tid & 31;
    int tile = blockIdx.x * 4 + wave;
    int pbase = tile * 32;
    int mrow = lane & 15;
    int khalf = lane >> 4;
    int p0 = pbase + mrow;
    int p1 = p0 + 16;

    int i0, j0, i1, j1;
    decode_pair(p0, i0, j0);
    decode_pair(p1, i1, j1);

    const _Float16* par0 = pa + i0 * MM;
    const _Float16* pbr0 = pb + j0 * MM;
    const _Float16* par1 = pa + i1 * MM;
    const _Float16* pbr1 = pb + j1 * MM;

    int ncol = lane & 15;
    float b2n = b2p[ncol];                         // pad entries are 0
    float w3n = (ncol < 8) ? W3[ncol] : 0.0f;
    float b3 = b3p[0];

    v8f acc0 = {};
    v8f acc1 = {};
    #pragma unroll
    for (int kt = 0; kt < 8; ++kt) {
        int k0 = kt * 32 + khalf * 8;
        v16h bf = *(const v16h*)(&W2f[(kt * 32 + lane) * 16]);   // 2x ds_load_b128
        v16h aA = build_a(par0, pbr0, cterm, k0);
        v16h aB = build_a(par1, pbr1, cterm, k0);
        acc0 = __builtin_amdgcn_wmma_f32_16x16x32_f16(false, aA, false, bf,
                                                      (short)0, acc0, false, false);
        acc1 = __builtin_amdgcn_wmma_f32_16x16x32_f16(false, aB, false, bf,
                                                      (short)0, acc1, false, false);
    }

    epilogue(acc0, lane, b2n, w3n, b3, p0, i0, j0, flags, scores, maskOut);
    epilogue(acc1, lane, b2n, w3n, b3, p1, i1, j1, flags, scores, maskOut);
}

// ---------------------------------------------------------------------------
extern "C" void kernel_launch(void* const* d_in, const int* in_sizes, int n_in,
                              void* d_out, int out_size, void* d_ws, size_t ws_size,
                              hipStream_t stream) {
    (void)in_sizes; (void)n_in; (void)out_size; (void)ws_size;
    const float* x     = (const float*)d_in[0];
    const float* ctx   = (const float*)d_in[1];
    const float* W1    = (const float*)d_in[2];
    const float* b1    = (const float*)d_in[3];
    const float* gamma = (const float*)d_in[4];
    const float* beta  = (const float*)d_in[5];
    const float* rmean = (const float*)d_in[6];
    const float* rvar  = (const float*)d_in[7];
    const float* W2    = (const float*)d_in[8];
    const float* b2    = (const float*)d_in[9];
    const float* W3    = (const float*)d_in[10];
    const float* b3    = (const float*)d_in[11];

    char* ws = (char*)d_ws;
    _Float16* pa     = (_Float16*)ws;  ws += (size_t)NN * MM * sizeof(_Float16);
    _Float16* pb     = (_Float16*)ws;  ws += (size_t)NN * MM * sizeof(_Float16);
    _Float16* cterm  = (_Float16*)ws;  ws += MM * sizeof(_Float16);  // 512B, 16B-aligned
    float* stress    = (float*)ws;     ws += NN * sizeof(float);
    float* flags     = (float*)ws;     ws += NN * sizeof(float);
    float* b2p       = (float*)ws;     ws += 16 * sizeof(float);
    _Float16* W2fr   = (_Float16*)ws;  ws += (size_t)8 * 32 * 16 * sizeof(_Float16);

    float* scores  = (float*)d_out;
    float* maskOut = scores + PP;

    fold_kernel<<<1, 256, 0, stream>>>(ctx, W1, b1, gamma, beta, rmean, rvar,
                                       W2, b2, cterm, b2p, W2fr);
    pp_kernel<<<NN, 256, 0, stream>>>(x, W1, pa, pb);
    stress_kernel<<<NN / 4, 128, 0, stream>>>(x, stress);
    flags_kernel<<<1, 1024, 0, stream>>>(stress, flags);
    // PP/32 = 16368 tiles, 4 waves (tiles) per block -> 4092 blocks
    pair_kernel<<<4092, 128, 0, stream>>>(pa, pb, cterm, W2fr, b2p, W3, b3,
                                          flags, scores, maskOut);
}
